// SelfAttention_33268816675404
// MI455X (gfx1250) — matrix operand order repaired and checked by reading, not compiled, not run on hardware
//
#include <hip/hip_runtime.h>
#include <math.h>

#define B_   4
#define N_   2048
#define D_   1024

#define LSTRIDE 136          // staged column stride in bf16 elems (128 + 8 pad -> bank-conflict-free)
#define LDS_BUF (64 * LSTRIDE)

typedef __attribute__((ext_vector_type(16))) __bf16 v16bf;
typedef __attribute__((ext_vector_type(8)))  float  v8f;
typedef __attribute__((ext_vector_type(4)))  int    v4i;

union Frag16 {
  v16bf v;
  uint4 q[2];
};

__device__ __forceinline__ unsigned short f2bf(float f) {
  unsigned int u = __float_as_uint(f);
  u += 0x7FFFu + ((u >> 16) & 1u);
  return (unsigned short)(u >> 16);
}

// ---- async global -> LDS copy (16B), ASYNCcnt-tracked; sync fallback ----
__device__ __forceinline__ void async_cp_b128(unsigned short* dst_lds,
                                              const unsigned short* src_global) {
#if __has_builtin(__builtin_amdgcn_global_load_async_to_lds_b128)
  __builtin_amdgcn_global_load_async_to_lds_b128(
      (__attribute__((address_space(1))) v4i*)src_global,
      (__attribute__((address_space(3))) v4i*)dst_lds, 0, 0);
#else
  *(uint4*)dst_lds = *(const uint4*)src_global;
#endif
}

__device__ __forceinline__ void async_wait_all() {
#if __has_builtin(__builtin_amdgcn_s_wait_asynccnt)
  __builtin_amdgcn_s_wait_asynccnt(0);
#elif __has_builtin(__builtin_amdgcn_global_load_async_to_lds_b128)
  asm volatile("s_wait_asynccnt 0x0" ::: "memory");
#endif
}

// ---------------- fp32 -> bf16 bulk convert (x) ----------------
__global__ __launch_bounds__(256) void k_cvt_x(const float* __restrict__ in,
                                               unsigned short* __restrict__ out) {
  int i = (blockIdx.x * 256 + threadIdx.x) * 4;
  float4 f = *(const float4*)(in + i);
  union { unsigned short s[4]; uint2 u; } o;
  o.s[0] = f2bf(f.x); o.s[1] = f2bf(f.y); o.s[2] = f2bf(f.z); o.s[3] = f2bf(f.w);
  *(uint2*)(out + i) = o.u;
}

// ------- weight transpose + convert: Wt[n][k] = bf16(W[k][n]) -------
__global__ __launch_bounds__(256) void k_cvt_wT(const float* __restrict__ Wq,
                                                const float* __restrict__ Wk,
                                                const float* __restrict__ Wv,
                                                unsigned short* __restrict__ Wt) {
  const float* W = (blockIdx.z == 0) ? Wq : (blockIdx.z == 1) ? Wk : Wv;
  unsigned short* o = Wt + (size_t)blockIdx.z * D_ * D_;
  __shared__ float tile[16][17];
  int kk = blockIdx.y * 16 + threadIdx.y;
  int nn = blockIdx.x * 16 + threadIdx.x;
  tile[threadIdx.y][threadIdx.x] = W[(size_t)kk * D_ + nn];
  __syncthreads();
  int on = blockIdx.x * 16 + threadIdx.y;
  int ok = blockIdx.y * 16 + threadIdx.x;
  o[(size_t)on * D_ + ok] = f2bf(tile[threadIdx.x][threadIdx.y]);
}

// ---------------- shared GEMM mainloop ----------------
// Per wave: 32 rows (two 16-row A fragments) x 64 cols (four B fragments).
// Every LDS B fragment feeds two WMMAs -> 8 wmma : 8 ds_load_b128 per 32-K step.
// B tile (64 cols x 128 K, bf16) is double-buffer staged in LDS via async DMA,
// shared by all 8 waves of the workgroup.
__device__ __forceinline__ void gemm_mainloop(const unsigned short* __restrict__ aptr0,
                                              const unsigned short* __restrict__ aptr1,
                                              const unsigned short* __restrict__ bbase,
                                              int K, int nbase,
                                              unsigned short* lds,
                                              v8f acc[2][4]) {
  const int tid  = threadIdx.x;
  const int lane = tid & 31;
  const int hi   = lane >> 4;
  const int l16  = lane & 15;

  auto stage = [&](int buf, int kc) {
    #pragma unroll
    for (int i = 0; i < 4; ++i) {
      const int p   = tid + i * 256;   // 1024 x 16B pieces = 64 cols x 128 elems
      const int col = p >> 4;
      const int kk  = (p & 15) * 8;
      async_cp_b128(lds + buf * LDS_BUF + col * LSTRIDE + kk,
                    bbase + (size_t)(nbase + col) * K + kc + kk);
    }
  };

  stage(0, 0);
  async_wait_all();
  __syncthreads();

  int buf = 0;
  for (int kc = 0; kc < K; kc += 128) {
    if (kc + 128 < K) stage(buf ^ 1, kc + 128);
    const unsigned short* lb = lds + buf * LDS_BUF;
    #pragma unroll
    for (int ks = 0; ks < 4; ++ks) {
      // load all four B fragments first so the 8-WMMA chain runs un-gated
      Frag16 bfr[4];
      #pragma unroll
      for (int j = 0; j < 4; ++j) {
        const unsigned short* bp = lb + (j * 16 + l16) * LSTRIDE + ks * 32 + hi * 16;
        bfr[j].q[0] = *(const uint4*)(bp);
        bfr[j].q[1] = *(const uint4*)(bp + 8);
      }
      const int ak = kc + ks * 32 + hi * 16;
      Frag16 a0, a1;
      a0.q[0] = *(const uint4*)(aptr0 + ak);
      a0.q[1] = *(const uint4*)(aptr0 + ak + 8);
      a1.q[0] = *(const uint4*)(aptr1 + ak);
      a1.q[1] = *(const uint4*)(aptr1 + ak + 8);
      #pragma unroll
      for (int j = 0; j < 4; ++j)
        acc[0][j] = __builtin_amdgcn_wmma_f32_16x16x32_bf16(false, a0.v, false, bfr[j].v,
                                                            (short)0, acc[0][j], false, false);
      #pragma unroll
      for (int j = 0; j < 4; ++j)
        acc[1][j] = __builtin_amdgcn_wmma_f32_16x16x32_bf16(false, a1.v, false, bfr[j].v,
                                                            (short)0, acc[1][j], false, false);
    }
    async_wait_all();
    __syncthreads();
    buf ^= 1;
  }
}

// ---------------- QKV projection GEMM ----------------
__global__ __launch_bounds__(256) void k_qkv(const unsigned short* __restrict__ xb,
                                             const unsigned short* __restrict__ Wt,
                                             const float* __restrict__ bq,
                                             const float* __restrict__ bk,
                                             const float* __restrict__ bv,
                                             unsigned short* __restrict__ Qb,
                                             unsigned short* __restrict__ Kb,
                                             unsigned short* __restrict__ Vt) {
  __shared__ unsigned short ldsB[2 * LDS_BUF];

  const int which = blockIdx.z % 3;
  const int b     = blockIdx.z / 3;
  const unsigned short* W = Wt + (size_t)which * D_ * D_;
  const float* bias = (which == 0) ? bq : (which == 1) ? bk : bv;

  const int w    = threadIdx.x >> 5;
  const int lane = threadIdx.x & 31;
  const int hi   = lane >> 4;
  const int l16  = lane & 15;

  const int mbase = blockIdx.x * 256 + w * 32;
  const int nbase = blockIdx.y * 64;

  const unsigned short* aptr0 = xb + ((size_t)b * N_ + (mbase + l16)) * D_;
  const unsigned short* aptr1 = aptr0 + (size_t)16 * D_;

  v8f acc[2][4];
  #pragma unroll
  for (int m = 0; m < 2; ++m)
    #pragma unroll
    for (int j = 0; j < 4; ++j)
      #pragma unroll
      for (int r = 0; r < 8; ++r) acc[m][j][r] = 0.f;

  gemm_mainloop(aptr0, aptr1, W, D_, nbase, ldsB, acc);

  #pragma unroll
  for (int m = 0; m < 2; ++m) {
    #pragma unroll
    for (int j = 0; j < 4; ++j) {
      const int col = nbase + j * 16 + l16;
      const float bs = bias[col];
      #pragma unroll
      for (int r = 0; r < 8; ++r) {
        const int row = mbase + m * 16 + r + hi * 8;
        const unsigned short val = f2bf(acc[m][j][r] + bs);
        if (which == 0)      Qb[((size_t)b * N_ + row) * D_ + col] = val;
        else if (which == 1) Kb[((size_t)b * N_ + row) * D_ + col] = val;
        else                 Vt[((size_t)b * D_ + col) * N_ + row] = val;
      }
    }
  }
}

// ---------------- scores GEMM: S = scale * Q . K^T ----------------
__global__ __launch_bounds__(256) void k_scores(const unsigned short* __restrict__ Qb,
                                                const unsigned short* __restrict__ Kb,
                                                float* __restrict__ S) {
  __shared__ unsigned short ldsB[2 * LDS_BUF];

  const int b    = blockIdx.z;
  const int w    = threadIdx.x >> 5;
  const int lane = threadIdx.x & 31;
  const int hi   = lane >> 4;
  const int l16  = lane & 15;

  const int qbase = blockIdx.x * 256 + w * 32;
  const int kbase = blockIdx.y * 64;

  const unsigned short* aptr0 = Qb + ((size_t)b * N_ + (qbase + l16)) * D_;
  const unsigned short* aptr1 = aptr0 + (size_t)16 * D_;
  const unsigned short* kbat  = Kb + (size_t)b * N_ * D_;

  v8f acc[2][4];
  #pragma unroll
  for (int m = 0; m < 2; ++m)
    #pragma unroll
    for (int j = 0; j < 4; ++j)
      #pragma unroll
      for (int r = 0; r < 8; ++r) acc[m][j][r] = 0.f;

  gemm_mainloop(aptr0, aptr1, kbat, D_, kbase, ldsB, acc);

  const float scale = 0.03125f;  // 1/sqrt(1024)
  #pragma unroll
  for (int m = 0; m < 2; ++m) {
    #pragma unroll
    for (int j = 0; j < 4; ++j) {
      const int col = kbase + j * 16 + l16;
      #pragma unroll
      for (int r = 0; r < 8; ++r) {
        const int row = qbase + m * 16 + r + hi * 8;
        S[((size_t)b * N_ + row) * N_ + col] = acc[m][j][r] * scale;
      }
    }
  }
}

// ---------------- row softmax over 2048, f32 -> bf16 ----------------
__global__ __launch_bounds__(256) void k_softmax(const float* __restrict__ S,
                                                 unsigned short* __restrict__ P) {
  const size_t row = blockIdx.x;
  const float* s = S + row * N_;
  unsigned short* p = P + row * N_;
  const int tid  = threadIdx.x;
  const int wid  = tid >> 5;
  const int lane = tid & 31;

  float v[8];
  float m = -3.4e38f;
  #pragma unroll
  for (int i = 0; i < 8; ++i) { v[i] = s[i * 256 + tid]; m = fmaxf(m, v[i]); }
  #pragma unroll
  for (int off = 16; off > 0; off >>= 1) m = fmaxf(m, __shfl_xor(m, off));

  __shared__ float red[8];
  if (lane == 0) red[wid] = m;
  __syncthreads();
  m = red[0];
  #pragma unroll
  for (int j = 1; j < 8; ++j) m = fmaxf(m, red[j]);

  float sum = 0.f;
  #pragma unroll
  for (int i = 0; i < 8; ++i) { v[i] = __expf(v[i] - m); sum += v[i]; }
  #pragma unroll
  for (int off = 16; off > 0; off >>= 1) sum += __shfl_xor(sum, off);
  __syncthreads();
  if (lane == 0) red[wid] = sum;
  __syncthreads();
  float tot = 0.f;
  #pragma unroll
  for (int j = 0; j < 8; ++j) tot += red[j];
  const float inv = 1.0f / tot;
  #pragma unroll
  for (int i = 0; i < 8; ++i) p[i * 256 + tid] = f2bf(v[i] * inv);
}

// ---------------- output GEMM: O = P . V ----------------
__global__ __launch_bounds__(256) void k_out(const unsigned short* __restrict__ P,
                                             const unsigned short* __restrict__ Vt,
                                             float* __restrict__ O) {
  __shared__ unsigned short ldsB[2 * LDS_BUF];

  const int b    = blockIdx.z;
  const int w    = threadIdx.x >> 5;
  const int lane = threadIdx.x & 31;
  const int hi   = lane >> 4;
  const int l16  = lane & 15;

  const int qbase = blockIdx.x * 256 + w * 32;
  const int vbase = blockIdx.y * 64;

  const unsigned short* aptr0 = P + ((size_t)b * N_ + (qbase + l16)) * N_;
  const unsigned short* aptr1 = aptr0 + (size_t)16 * N_;
  const unsigned short* vbat  = Vt + (size_t)b * D_ * N_;

  v8f acc[2][4];
  #pragma unroll
  for (int m = 0; m < 2; ++m)
    #pragma unroll
    for (int j = 0; j < 4; ++j)
      #pragma unroll
      for (int r = 0; r < 8; ++r) acc[m][j][r] = 0.f;

  gemm_mainloop(aptr0, aptr1, vbat, N_, vbase, ldsB, acc);

  #pragma unroll
  for (int m = 0; m < 2; ++m) {
    #pragma unroll
    for (int j = 0; j < 4; ++j) {
      const int col = vbase + j * 16 + l16;
      #pragma unroll
      for (int r = 0; r < 8; ++r) {
        const int row = qbase + m * 16 + r + hi * 8;
        O[((size_t)b * N_ + row) * D_ + col] = acc[m][j][r];
      }
    }
  }
}

extern "C" void kernel_launch(void* const* d_in, const int* in_sizes, int n_in,
                              void* d_out, int out_size, void* d_ws, size_t ws_size,
                              hipStream_t stream) {
  (void)in_sizes; (void)n_in; (void)out_size; (void)ws_size;
  const float* x  = (const float*)d_in[0];
  const float* Wq = (const float*)d_in[1];
  const float* bq = (const float*)d_in[2];
  const float* Wk = (const float*)d_in[3];
  const float* bk = (const float*)d_in[4];
  const float* Wv = (const float*)d_in[5];
  const float* bv = (const float*)d_in[6];
  float* out = (float*)d_out;

  // workspace carve-up (bf16 stored as ushort)
  unsigned short* wsu = (unsigned short*)d_ws;
  size_t o = 0;
  unsigned short* xb = wsu + o; o += (size_t)B_ * N_ * D_;      // x in bf16
  unsigned short* Wt = wsu + o; o += (size_t)3 * D_ * D_;       // Wq/Wk/Wv transposed bf16
  unsigned short* Qb = wsu + o; o += (size_t)B_ * N_ * D_;      // Q bf16 row-major
  unsigned short* Kb = wsu + o; o += (size_t)B_ * N_ * D_;      // K bf16 row-major
  unsigned short* Vt = wsu + o; o += (size_t)B_ * N_ * D_;      // V bf16 transposed [b][v][tok]
  unsigned short* P  = wsu + o; o += (size_t)B_ * N_ * N_;      // softmax probs bf16
  float* S = (float*)(wsu + o);                                 // raw scores f32

  k_cvt_x   <<<(B_ * N_ * D_) / 1024, 256, 0, stream>>>(x, xb);
  k_cvt_wT  <<<dim3(64, 64, 3), dim3(16, 16), 0, stream>>>(Wq, Wk, Wv, Wt);
  k_qkv     <<<dim3(8, 16, 12), 256, 0, stream>>>(xb, Wt, bq, bk, bv, Qb, Kb, Vt);
  k_scores  <<<dim3(8, 32, 4), 256, 0, stream>>>(Qb, Kb, S);
  k_softmax <<<B_ * N_, 256, 0, stream>>>(S, P);
  k_out     <<<dim3(8, 16, 4), 256, 0, stream>>>(P, Vt, out);
}
